// VisionTransformer_77189152244476
// MI455X (gfx1250) — matrix operand order repaired and checked by reading, not compiled, not run on hardware
//
#include <hip/hip_runtime.h>
#include <math.h>

// ---------------------------------------------------------------------------
// Types for CDNA5 WMMA (gfx1250, wave32)
// ---------------------------------------------------------------------------
typedef __attribute__((ext_vector_type(16))) __bf16 v16bf;
typedef __attribute__((ext_vector_type(8)))  __bf16 v8bf;
typedef __attribute__((ext_vector_type(4)))  __bf16 v4bf;
typedef __attribute__((ext_vector_type(8)))  float  v8f;

// ---------------------------------------------------------------------------
// Problem constants (from the reference)
// ---------------------------------------------------------------------------
#define IMG     512
#define PSZ     16
#define CIN     3
#define DMODEL  768
#define NHEAD   8
#define DHEAD   96          // DMODEL / NHEAD
#define MLPH    3072
#define NCLS    19
#define NTOK    1024        // (512/16)^2 patches per image
#define BSZ     2
#define NROWS   (BSZ * NTOK)   // 2048 token rows total
#define LN_EPS  1e-5f

// ---------------------------------------------------------------------------
// GEMM tiling: 128x128x32 block tile, 256 threads = 8 waves,
// each wave computes a 32x64 sub-tile = 2x4 WMMA 16x16 accumulators.
// Both LDS tiles use the SAME layout [outer dim][K], padded so every
// per-lane fragment chunk is a 16-byte aligned ds_load_b128.
// Global->LDS staging is register-buffered and software-pipelined one
// k-step ahead so global latency hides behind the WMMA pipe.
// ---------------------------------------------------------------------------
#define BM 128
#define BN 128
#define BK 32
#define TILE_LD 40   // halves per row: 80 B row stride, 16-B aligned chunks

template<bool TRANS_B, bool FULL>
__global__ __launch_bounds__(256)
void gemm_wmma_bf16(const float* __restrict__ A, const float* __restrict__ B,
                    float* __restrict__ C,
                    int M, int N, int K, int lda, int ldb, int ldc,
                    long long sAb, long long sAh, long long sBb, long long sBh,
                    long long sCb, long long sCh, int Hdim,
                    const float* __restrict__ bias,
                    const float* __restrict__ res1, int res1Mod,
                    const float* __restrict__ res2,
                    float alpha, int act)
{
    __shared__ __bf16 As [BM * TILE_LD];   // [m][k]
    __shared__ __bf16 BsT[BN * TILE_LD];   // [n][k]  (B transposed into LDS)

    // batched (b, h) offsets for attention GEMMs; strides are 0 when unbatched
    const int z  = blockIdx.z;
    const int bb = z / Hdim;
    const int hh = z % Hdim;
    A += bb * sAb + hh * sAh;
    B += bb * sBb + hh * sBh;
    C += bb * sCb + hh * sCh;

    const int t    = threadIdx.x;
    const int lane = t & 31;
    const int wv   = t >> 5;
    const int hi   = (lane >> 4) & 1;   // lane half (0: lanes 0-15, 1: 16-31)
    const int l16  = lane & 15;

    const int m0 = blockIdx.y * BM;
    const int n0 = blockIdx.x * BN;

    const int wRow = (wv >> 1) * 32;    // 4 wave-rows of 32
    const int wCol = (wv & 1)  * 64;    // 2 wave-cols of 64

    v8f acc[2][4] = {};                 // zero-init f32 accumulators

    // ---- register staging buffers (distinct regs -> loads stay in flight) --
    float4 ra[4], rb[4];                // FULL path
    float  sa[16], sb[16];              // guarded path

    // gather one k-slab of A and B into registers (no LDS, no waits forced)
    auto load_regs = [&](int k0) {
        if (FULL) {
            #pragma unroll
            for (int i = 0; i < 4; ++i) {
                int e = t + i * 256;
                int r = e >> 3, c4 = (e & 7) * 4;
                ra[i] = *(const float4*)&A[(size_t)(m0 + r) * lda + (k0 + c4)];
            }
            if (TRANS_B) {
                #pragma unroll
                for (int i = 0; i < 4; ++i) {
                    int e = t + i * 256;
                    int n = e >> 3, c4 = (e & 7) * 4;
                    rb[i] = *(const float4*)&B[(size_t)(n0 + n) * ldb + (k0 + c4)];
                }
            } else {
                #pragma unroll
                for (int i = 0; i < 4; ++i) {
                    int e = t + i * 256;
                    int r = e >> 5, c4 = (e & 31) * 4;
                    rb[i] = *(const float4*)&B[(size_t)(k0 + r) * ldb + (n0 + c4)];
                }
            }
        } else {
            // clamped-index unconditional loads (no exec-divergent branches)
            #pragma unroll
            for (int i = 0; i < 16; ++i) {
                int e = t + i * 256;
                int r = e >> 5, c = e & 31;
                int m  = m0 + r;
                int mc = (m < M) ? m : (M - 1);
                float v = A[(size_t)mc * lda + (k0 + c)];
                sa[i] = (m < M) ? v : 0.0f;
            }
            if (TRANS_B) {
                #pragma unroll
                for (int i = 0; i < 16; ++i) {
                    int e = t + i * 256;
                    int n = e >> 5, c = e & 31;
                    int gn  = n0 + n;
                    int gnc = (gn < N) ? gn : (N - 1);
                    float v = B[(size_t)gnc * ldb + (k0 + c)];
                    sb[i] = (gn < N) ? v : 0.0f;
                }
            } else {
                #pragma unroll
                for (int i = 0; i < 16; ++i) {
                    int e = t + i * 256;
                    int r = e >> 7, c = e & 127;
                    int gn  = n0 + c;
                    int gnc = (gn < N) ? gn : (N - 1);
                    float v = B[(size_t)(k0 + r) * ldb + gnc];
                    sb[i] = (gn < N) ? v : 0.0f;
                }
            }
        }
    };

    // convert staged registers to bf16 and store into LDS tiles
    auto store_lds = [&]() {
        if (FULL) {
            #pragma unroll
            for (int i = 0; i < 4; ++i) {
                int e = t + i * 256;
                int r = e >> 3, c4 = (e & 7) * 4;
                v4bf h = { (__bf16)ra[i].x, (__bf16)ra[i].y,
                           (__bf16)ra[i].z, (__bf16)ra[i].w };
                *(v4bf*)&As[r * TILE_LD + c4] = h;
            }
            if (TRANS_B) {
                #pragma unroll
                for (int i = 0; i < 4; ++i) {
                    int e = t + i * 256;
                    int n = e >> 3, c4 = (e & 7) * 4;
                    v4bf h = { (__bf16)rb[i].x, (__bf16)rb[i].y,
                               (__bf16)rb[i].z, (__bf16)rb[i].w };
                    *(v4bf*)&BsT[n * TILE_LD + c4] = h;
                }
            } else {
                #pragma unroll
                for (int i = 0; i < 4; ++i) {
                    int e = t + i * 256;
                    int r = e >> 5, c4 = (e & 31) * 4;
                    BsT[(c4 + 0) * TILE_LD + r] = (__bf16)rb[i].x;
                    BsT[(c4 + 1) * TILE_LD + r] = (__bf16)rb[i].y;
                    BsT[(c4 + 2) * TILE_LD + r] = (__bf16)rb[i].z;
                    BsT[(c4 + 3) * TILE_LD + r] = (__bf16)rb[i].w;
                }
            }
        } else {
            #pragma unroll
            for (int i = 0; i < 16; ++i) {
                int e = t + i * 256;
                int r = e >> 5, c = e & 31;
                As[r * TILE_LD + c] = (__bf16)sa[i];
            }
            if (TRANS_B) {
                #pragma unroll
                for (int i = 0; i < 16; ++i) {
                    int e = t + i * 256;
                    int n = e >> 5, c = e & 31;
                    BsT[n * TILE_LD + c] = (__bf16)sb[i];
                }
            } else {
                #pragma unroll
                for (int i = 0; i < 16; ++i) {
                    int e = t + i * 256;
                    int r = e >> 7, c = e & 127;
                    BsT[c * TILE_LD + r] = (__bf16)sb[i];
                }
            }
        }
    };

    // ---- software pipeline: regs hold tile k+1 while WMMA consumes tile k --
    load_regs(0);
    for (int k0 = 0; k0 < K; k0 += BK) {
        store_lds();                      // waits for tile-k loads land here
        __syncthreads();
        if (k0 + BK < K) load_regs(k0 + BK);   // issue next-tile loads now

        // ---- fragments via 16-B LDS chunk loads (ISA 7.12.2 layouts) ----
        // A (16x32): lane holds row M=lane%16; lane-half selects K sub-bands:
        //   halves 0-7 -> K = hi*8 + 0..7 ;  halves 8-15 -> K = 16+hi*8+0..7
        v16bf afr[2];
        #pragma unroll
        for (int mt = 0; mt < 2; ++mt) {
            const __bf16* rowp = &As[(wRow + mt * 16 + l16) * TILE_LD];
            v8bf a0 = *(const v8bf*)(rowp + hi * 8);
            v8bf a1 = *(const v8bf*)(rowp + 16 + hi * 8);
            afr[mt] = __builtin_shufflevector(a0, a1,
                0, 1, 2, 3, 4, 5, 6, 7, 8, 9, 10, 11, 12, 13, 14, 15);
        }
        // B (32x16): lane holds col N=lane%16; same K sub-bands as A:
        //   halves 0-7 -> K = hi*8 + 0..7 ;  halves 8-15 -> K = 16+hi*8+0..7
        v16bf bfr[4];
        #pragma unroll
        for (int nt = 0; nt < 4; ++nt) {
            const __bf16* colp = &BsT[(wCol + nt * 16 + l16) * TILE_LD];
            v8bf b0 = *(const v8bf*)(colp + hi * 8);
            v8bf b1 = *(const v8bf*)(colp + 16 + hi * 8);
            bfr[nt] = __builtin_shufflevector(b0, b1,
                0, 1, 2, 3, 4, 5, 6, 7, 8, 9, 10, 11, 12, 13, 14, 15);
        }

        #pragma unroll
        for (int mt = 0; mt < 2; ++mt)
            #pragma unroll
            for (int nt = 0; nt < 4; ++nt)
                acc[mt][nt] = __builtin_amdgcn_wmma_f32_16x16x32_bf16(
                    false, afr[mt], false, bfr[nt],
                    (short)0, acc[mt][nt], false, false);
        __syncthreads();                  // LDS free before next store_lds
    }

    // ---- epilogue: alpha, bias, exact GELU, up to two residual adds ----
    // C/D layout: VGPR r holds M = r + 8*laneHalf, N = lane%16
    #pragma unroll
    for (int mt = 0; mt < 2; ++mt) {
        #pragma unroll
        for (int nt = 0; nt < 4; ++nt) {
            int n = n0 + wCol + nt * 16 + l16;
            if (n >= N) continue;
            #pragma unroll
            for (int r = 0; r < 8; ++r) {
                int m = m0 + wRow + mt * 16 + r + hi * 8;
                if (!FULL && m >= M) continue;
                float v = alpha * acc[mt][nt][r];
                if (bias) v += bias[n];
                if (act == 1) v = 0.5f * v * (1.0f + erff(v * 0.70710678118654752f));
                if (res1) {
                    int rm = res1Mod ? (m % res1Mod) : m;
                    v += res1[(size_t)rm * ldc + n];
                }
                if (res2) v += res2[(size_t)m * ldc + n];
                C[(size_t)m * ldc + n] = v;
            }
        }
    }
}

// ---------------------------------------------------------------------------
// im2col for the stride-16 patch conv: out[b*1024+n][k], k = c*256+py*16+px
// ---------------------------------------------------------------------------
__global__ __launch_bounds__(256)
void patchify_kernel(const float* __restrict__ x, float* __restrict__ out)
{
    int idx = blockIdx.x * 256 + threadIdx.x;
    if (idx >= NROWS * DMODEL) return;
    int row = idx / DMODEL, k = idx % DMODEL;
    int b = row / NTOK, n = row % NTOK;
    int ph = n >> 5, pw = n & 31;
    int c = k >> 8, r = k & 255;
    int py = r >> 4, px = r & 15;
    out[idx] = x[((size_t)b * CIN + c) * IMG * IMG
                 + (size_t)(ph * PSZ + py) * IMG + (pw * PSZ + px)];
}

// ---------------------------------------------------------------------------
// Row LayerNorm (rows x cols), 256 threads/row, wave32 shuffle reductions
// ---------------------------------------------------------------------------
__global__ __launch_bounds__(256)
void layernorm_kernel(const float* __restrict__ x, float* __restrict__ y,
                      const float* __restrict__ w, const float* __restrict__ b,
                      int cols)
{
    __shared__ float s1[8], s2[8];
    const float* xp = x + (size_t)blockIdx.x * cols;
    float* yp = y + (size_t)blockIdx.x * cols;
    int t = threadIdx.x, lane = t & 31, wv = t >> 5;

    float sum = 0.f, sq = 0.f;
    for (int c = t; c < cols; c += 256) { float v = xp[c]; sum += v; sq += v * v; }
    for (int off = 16; off > 0; off >>= 1) {
        sum += __shfl_down(sum, off, 32);
        sq  += __shfl_down(sq,  off, 32);
    }
    if (lane == 0) { s1[wv] = sum; s2[wv] = sq; }
    __syncthreads();
    sum = 0.f; sq = 0.f;
    #pragma unroll
    for (int i = 0; i < 8; ++i) { sum += s1[i]; sq += s2[i]; }
    float mean = sum / cols;
    float var  = sq / cols - mean * mean;
    float inv  = rsqrtf(var + LN_EPS);
    for (int c = t; c < cols; c += 256)
        yp[c] = (xp[c] - mean) * inv * w[c] + b[c];
}

// ---------------------------------------------------------------------------
// In-place row softmax over `cols` elements per block row
// ---------------------------------------------------------------------------
__global__ __launch_bounds__(256)
void softmax_kernel(float* __restrict__ s, int cols)
{
    __shared__ float red[8];
    float* p = s + (size_t)blockIdx.x * cols;
    int t = threadIdx.x, lane = t & 31, wv = t >> 5;

    float mx = -3.402823466e38f;
    for (int c = t; c < cols; c += 256) mx = fmaxf(mx, p[c]);
    for (int off = 16; off > 0; off >>= 1) mx = fmaxf(mx, __shfl_down(mx, off, 32));
    if (lane == 0) red[wv] = mx;
    __syncthreads();
    mx = red[0];
    #pragma unroll
    for (int i = 1; i < 8; ++i) mx = fmaxf(mx, red[i]);

    float sum = 0.f;
    for (int c = t; c < cols; c += 256) {
        float e = __expf(p[c] - mx);
        p[c] = e;
        sum += e;
    }
    for (int off = 16; off > 0; off >>= 1) sum += __shfl_down(sum, off, 32);
    __syncthreads();
    if (lane == 0) red[wv] = sum;
    __syncthreads();
    sum = 0.f;
    #pragma unroll
    for (int i = 0; i < 8; ++i) sum += red[i];
    float inv = 1.f / sum;
    for (int c = t; c < cols; c += 256) p[c] *= inv;
}

// ---------------------------------------------------------------------------
// 16x bilinear upsample, half-pixel centers: cls[b][n=py*32+px][c] ->
// out[b][c][512][512]
// ---------------------------------------------------------------------------
__global__ __launch_bounds__(256)
void upsample_kernel(const float* __restrict__ cls, float* __restrict__ out,
                     size_t total)
{
    size_t idx = (size_t)blockIdx.x * 256 + threadIdx.x;
    if (idx >= total) return;
    int xo = (int)(idx % IMG); size_t tt = idx / IMG;
    int yo = (int)(tt % IMG);  tt /= IMG;
    int c  = (int)(tt % NCLS);
    int b  = (int)(tt / NCLS);

    float fy = (yo + 0.5f) * (1.0f / 16.0f) - 0.5f;
    float fx = (xo + 0.5f) * (1.0f / 16.0f) - 0.5f;
    int y0 = (int)floorf(fy); float wy = fy - (float)y0;
    int x0 = (int)floorf(fx); float wx = fx - (float)x0;
    int y1 = y0 + 1, x1 = x0 + 1;
    y0 = min(max(y0, 0), 31); y1 = min(max(y1, 0), 31);
    x0 = min(max(x0, 0), 31); x1 = min(max(x1, 0), 31);

    auto V = [&](int py, int px) {
        return cls[((size_t)b * NTOK + py * 32 + px) * NCLS + c];
    };
    float v = (1.f - wy) * ((1.f - wx) * V(y0, x0) + wx * V(y0, x1))
            +        wy  * ((1.f - wx) * V(y1, x0) + wx * V(y1, x1));
    out[idx] = v;
}

// ---------------------------------------------------------------------------
// Host-side orchestration
// ---------------------------------------------------------------------------
static void launch_gemm(hipStream_t st, bool transB,
                        const float* A, const float* B, float* C,
                        int M, int N, int K, int lda, int ldb, int ldc,
                        const float* bias, const float* res1, int res1Mod,
                        const float* res2, float alpha, int act,
                        int batch = 1, int Hdim = 1,
                        long long sAb = 0, long long sAh = 0,
                        long long sBb = 0, long long sBh = 0,
                        long long sCb = 0, long long sCh = 0)
{
    dim3 grid((N + BN - 1) / BN, (M + BM - 1) / BM, batch);
    const bool full = (M % BM == 0) && (N % BN == 0);
    if (transB) {
        if (full)
            gemm_wmma_bf16<true, true><<<grid, 256, 0, st>>>(A, B, C, M, N, K,
                lda, ldb, ldc, sAb, sAh, sBb, sBh, sCb, sCh, Hdim,
                bias, res1, res1Mod, res2, alpha, act);
        else
            gemm_wmma_bf16<true, false><<<grid, 256, 0, st>>>(A, B, C, M, N, K,
                lda, ldb, ldc, sAb, sAh, sBb, sBh, sCb, sCh, Hdim,
                bias, res1, res1Mod, res2, alpha, act);
    } else {
        if (full)
            gemm_wmma_bf16<false, true><<<grid, 256, 0, st>>>(A, B, C, M, N, K,
                lda, ldb, ldc, sAb, sAh, sBb, sBh, sCb, sCh, Hdim,
                bias, res1, res1Mod, res2, alpha, act);
        else
            gemm_wmma_bf16<false, false><<<grid, 256, 0, st>>>(A, B, C, M, N, K,
                lda, ldb, ldc, sAb, sAh, sBb, sBh, sCb, sCh, Hdim,
                bias, res1, res1Mod, res2, alpha, act);
    }
}

struct BlkP {
    const float *n1_w, *n1_b, *qkv_w, *proj_w, *proj_b;
    const float *n2_w, *n2_b, *fc1_w, *fc1_b, *fc2_w, *fc2_b;
};

static void run_block(hipStream_t st, const BlkP& p,
                      const float* xin, float* xout, const float* skip,
                      float* lnbuf, float* qkv, float* scores,
                      float* o, float* xattn, float* mlph)
{
    const float scale = 0.1020620726159657f;  // 1/sqrt(96)

    // h = LN(x)
    layernorm_kernel<<<NROWS, 256, 0, st>>>(xin, lnbuf, p.n1_w, p.n1_b, DMODEL);
    // qkv = h @ qkv_w           [2048 x 2304]
    launch_gemm(st, false, lnbuf, p.qkv_w, qkv,
                NROWS, 3 * DMODEL, DMODEL, DMODEL, 3 * DMODEL, 3 * DMODEL,
                nullptr, nullptr, 0, nullptr, 1.f, 0);
    // scores = scale * q @ k^T  per (b,h): [1024 x 1024], K = 96
    launch_gemm(st, true, qkv, qkv + DMODEL, scores,
                NTOK, NTOK, DHEAD, 3 * DMODEL, 3 * DMODEL, NTOK,
                nullptr, nullptr, 0, nullptr, scale, 0,
                BSZ * NHEAD, NHEAD,
                (long long)NTOK * 3 * DMODEL, DHEAD,
                (long long)NTOK * 3 * DMODEL, DHEAD,
                (long long)NHEAD * NTOK * NTOK, (long long)NTOK * NTOK);
    softmax_kernel<<<BSZ * NHEAD * NTOK, 256, 0, st>>>(scores, NTOK);
    // o[b, n, h*96 + d] = A @ v  per (b,h): [1024 x 96], K = 1024
    launch_gemm(st, false, scores, qkv + 2 * DMODEL, o,
                NTOK, DHEAD, NTOK, NTOK, 3 * DMODEL, DMODEL,
                nullptr, nullptr, 0, nullptr, 1.f, 0,
                BSZ * NHEAD, NHEAD,
                (long long)NHEAD * NTOK * NTOK, (long long)NTOK * NTOK,
                (long long)NTOK * 3 * DMODEL, DHEAD,
                (long long)NTOK * DMODEL, DHEAD);
    // xattn = x + o @ proj_w + proj_b
    launch_gemm(st, false, o, p.proj_w, xattn,
                NROWS, DMODEL, DMODEL, DMODEL, DMODEL, DMODEL,
                p.proj_b, xin, 0, nullptr, 1.f, 0);
    // MLP
    layernorm_kernel<<<NROWS, 256, 0, st>>>(xattn, lnbuf, p.n2_w, p.n2_b, DMODEL);
    launch_gemm(st, false, lnbuf, p.fc1_w, mlph,
                NROWS, MLPH, DMODEL, DMODEL, MLPH, MLPH,
                p.fc1_b, nullptr, 0, nullptr, 1.f, /*gelu*/1);
    // xout = xattn + mlp(h) [+ skip]   (decoder skip fused into epilogue)
    launch_gemm(st, false, mlph, p.fc2_w, xout,
                NROWS, DMODEL, MLPH, MLPH, DMODEL, DMODEL,
                p.fc2_b, xattn, 0, skip, 1.f, 0);
}

extern "C" void kernel_launch(void* const* d_in, const int* in_sizes, int n_in,
                              void* d_out, int out_size, void* d_ws, size_t ws_size,
                              hipStream_t stream)
{
    (void)in_sizes; (void)n_in; (void)out_size; (void)ws_size;

    // setup_inputs() insertion order: x, then params (conv_w, conv_b, pos,
    // cls_w, cls_b, blocks[0..7] with each block's insertion-order leaves)
    const float* x      = (const float*)d_in[0];
    const float* conv_w = (const float*)d_in[1];
    const float* conv_b = (const float*)d_in[2];
    const float* pos    = (const float*)d_in[3];
    const float* cls_w  = (const float*)d_in[4];
    const float* cls_b  = (const float*)d_in[5];
    BlkP bp[8];
    for (int i = 0; i < 8; ++i) {
        const int base = 6 + i * 11;
        bp[i].n1_w   = (const float*)d_in[base + 0];
        bp[i].n1_b   = (const float*)d_in[base + 1];
        bp[i].qkv_w  = (const float*)d_in[base + 2];
        bp[i].proj_w = (const float*)d_in[base + 3];
        bp[i].proj_b = (const float*)d_in[base + 4];
        bp[i].n2_w   = (const float*)d_in[base + 5];
        bp[i].n2_b   = (const float*)d_in[base + 6];
        bp[i].fc1_w  = (const float*)d_in[base + 7];
        bp[i].fc1_b  = (const float*)d_in[base + 8];
        bp[i].fc2_w  = (const float*)d_in[base + 9];
        bp[i].fc2_b  = (const float*)d_in[base + 10];
    }

    // Workspace carve-up (floats)
    float* w = (float*)d_ws;
    size_t off = 0;
    auto alloc = [&](size_t n) { float* p = w + off; off += n; return p; };
    float* patches = alloc((size_t)NROWS * DMODEL);
    float* z       = alloc((size_t)NROWS * DMODEL);
    float* lnbuf   = alloc((size_t)NROWS * DMODEL);
    float* qkv     = alloc((size_t)NROWS * 3 * DMODEL);
    float* scores  = alloc((size_t)BSZ * NHEAD * NTOK * NTOK);
    float* o       = alloc((size_t)NROWS * DMODEL);
    float* xattn   = alloc((size_t)NROWS * DMODEL);
    float* mlph    = alloc((size_t)NROWS * MLPH);
    float* x1      = alloc((size_t)NROWS * DMODEL);
    float* x2      = alloc((size_t)NROWS * DMODEL);
    float* x3      = alloc((size_t)NROWS * DMODEL);
    float* yA      = alloc((size_t)NROWS * DMODEL);
    float* yB      = alloc((size_t)NROWS * DMODEL);
    float* clsbuf  = alloc((size_t)NROWS * NCLS);

    // Patch embed: im2col + GEMM vs conv_w^T, fused (+conv_b, +pos broadcast)
    patchify_kernel<<<(NROWS * DMODEL + 255) / 256, 256, 0, stream>>>(x, patches);
    launch_gemm(stream, true, patches, conv_w, z,
                NROWS, DMODEL, DMODEL, DMODEL, DMODEL, DMODEL,
                conv_b, pos, /*res1Mod=*/NTOK, nullptr, 1.f, 0);

    // Encoder fan-out (blocks 0-2 from z); block 3 is dead code in the
    // reference forward (_x4 unused) and is skipped.
    run_block(stream, bp[0], z, x1, nullptr, lnbuf, qkv, scores, o, xattn, mlph);
    run_block(stream, bp[1], z, x2, nullptr, lnbuf, qkv, scores, o, xattn, mlph);
    run_block(stream, bp[2], z, x3, nullptr, lnbuf, qkv, scores, o, xattn, mlph);
    // Decoder chain with fused skip additions
    run_block(stream, bp[4], z,  yA, nullptr, lnbuf, qkv, scores, o, xattn, mlph);
    run_block(stream, bp[5], yA, yB, x3,      lnbuf, qkv, scores, o, xattn, mlph);
    run_block(stream, bp[6], yB, yA, x2,      lnbuf, qkv, scores, o, xattn, mlph);
    run_block(stream, bp[7], yA, yB, x1,      lnbuf, qkv, scores, o, xattn, mlph);

    // Classifier head [2048 x 19]
    launch_gemm(stream, false, yB, cls_w, clsbuf,
                NROWS, NCLS, DMODEL, DMODEL, NCLS, NCLS,
                cls_b, nullptr, 0, nullptr, 1.f, 0);

    // Bilinear 16x upsample to [2, 19, 512, 512]
    size_t total = (size_t)BSZ * NCLS * IMG * IMG;
    upsample_kernel<<<(unsigned)((total + 255) / 256), 256, 0, stream>>>(
        clsbuf, (float*)d_out, total);
}
